// MemristorGNN_63127429317286
// MI455X (gfx1250) — matrix operand order repaired and checked by reading, not compiled, not run on hardware
//
#include <hip/hip_runtime.h>
#include <hip/hip_bf16.h>
#include <math.h>

// ---------------------------------------------------------------------------
// MemristorGNN for MI455X (gfx1250, wave32, WMMA).
//
// Factorization: msg_e[o] = sum_k h_e[k] * G[src, k*dout+o] + G[src, Cb+o]
// where G[n, k*dout+o] = sum_i h[n,i] * w2[k, i*dout+o]   (node-level GEMM)
//       G[n, Cb+o]     = sum_i h[n,i] * b2[i*dout+o]      (bias fold)
//       G[n, Cr+o]     = sum_i h[n,i] * root[i,o]         (root fold)
// G per layer: 10000 x 1088 f32 = 43.5MB -> L2-resident on MI455X (192MB L2).
// GEMM runs on v_wmma_f32_16x16x32_bf16 (K=32 in one op, f32 accumulate);
// each wave keeps its A fragment in registers and sweeps 4 N-tiles (4 WMMAs).
// ---------------------------------------------------------------------------

typedef __attribute__((ext_vector_type(16))) __bf16 v16bf;
typedef __attribute__((ext_vector_type(8)))  float  v8f;

#define N_NODES 10000
#define N_EDGES 100000
#define NTB 4   // N-tiles per wave in the WMMA GEMM (A-register reuse)

static __device__ __forceinline__ unsigned short f2bf(float f) {
    unsigned u = __float_as_uint(f);
    unsigned r = ((u >> 16) & 1u) + 0x7fffu;   // round-to-nearest-even
    return (unsigned short)((u + r) >> 16);
}

// ---------------- zero fill ----------------
__global__ void k_clear(float* p, int n) {
    for (int i = blockIdx.x * blockDim.x + threadIdx.x; i < n;
         i += gridDim.x * blockDim.x) p[i] = 0.0f;
}

// ---------------- degree (once) ----------------
__global__ void k_deg(const int* ei, float* deg, int E) {
    int e = blockIdx.x * blockDim.x + threadIdx.x;
    if (e < E) atomicAdd(&deg[ei[E + e]], 1.0f);
}

// ---------------- voltage encoder + input features ----------------
// vemb = (relu(v*vw1+vb1)) @ vw2 + vb2 ; h0 = [x(10), vemb(8)] -> bf16 pad 32
__global__ void k_venc(const float* x, const float* vw1, const float* vb1,
                       const float* vw2, const float* vb2,
                       float* vemb, unsigned short* hbf, int N) {
    int n = blockIdx.x * blockDim.x + threadIdx.x;
    if (n >= N) return;
    float v = x[n * 10 + 7];                 // x[:, -3:-2]
    float t1[8];
    #pragma unroll
    for (int j = 0; j < 8; ++j) { float t = v * vw1[j] + vb1[j]; t1[j] = t > 0.f ? t : 0.f; }
    unsigned short* hb = hbf + (size_t)n * 32;
    #pragma unroll
    for (int k = 0; k < 8; ++k) {
        float a = vb2[k];
        #pragma unroll
        for (int j = 0; j < 8; ++j) a += t1[j] * vw2[j * 8 + k];
        vemb[n * 8 + k] = a;
        hb[10 + k] = f2bf(a);
    }
    #pragma unroll
    for (int c = 0; c < 10; ++c) hb[c] = f2bf(x[n * 10 + c]);
    #pragma unroll
    for (int c = 18; c < 32; ++c) hb[c] = 0;
}

// ---------------- pack B into WMMA-ready per-lane bf16 layout ----------------
// Bp[tile][lane][j] : row i = (j&7) + (lane>=16 ? 8:0) + (j>=8 ? 16:0), col c = tile*16 + (lane&15)
// Bsrc[i][c]: c<32*dout -> w2[kk*din*dout + i*dout + o]  (kk=c/dout, o=c%dout)
//             c<33*dout -> b2[i*dout+o] ; c<34*dout -> root[i*dout+o] ; else 0
__global__ void k_pack(const float* w2, const float* b2, const float* root,
                       int din, int dout, int NT, unsigned short* Bp) {
    int tid = blockIdx.x * blockDim.x + threadIdx.x;
    if (tid >= NT * 32) return;
    int t = tid >> 5, l = tid & 31;
    unsigned short* out = Bp + (size_t)tid * 16;
    int c = t * 16 + (l & 15);
    #pragma unroll
    for (int j = 0; j < 16; ++j) {
        int i = (j & 7) + ((l >= 16) ? 8 : 0) + ((j >= 8) ? 16 : 0);
        float v = 0.0f;
        if (i < din && c < 34 * dout) {
            if (c < 32 * dout) {
                int kk = c / dout, o = c - kk * dout;
                v = w2[(size_t)kk * din * dout + i * dout + o];
            } else if (c < 33 * dout) {
                v = b2[i * dout + (c - 32 * dout)];
            } else {
                v = root[i * dout + (c - 33 * dout)];
            }
        }
        out[j] = f2bf(v);
    }
}

// ---------------- WMMA GEMM: G[MxNG] = A[Mx32](bf16) @ B ----------------
// one wave per (16 x NTB*16) output strip; A fragment stays in registers and
// is reused across NTB consecutive N-tiles (4 v_wmma per wave).
__global__ __launch_bounds__(256) void k_gemm(const unsigned short* A,
                                              const unsigned short* Bp,
                                              float* G, int MT, int NT, int NG) {
    int NGP  = (NT + NTB - 1) / NTB;
    int wid  = blockIdx.x * (blockDim.x >> 5) + (threadIdx.x >> 5);
    int lane = threadIdx.x & 31;
    if (wid >= MT * NGP) return;              // wave-uniform exit (EXEC all-1 for WMMA)
    int tm  = wid / NGP;
    int tn0 = (wid - tm * NGP) * NTB;
    int half = lane >> 4;

    // A: lane L(<16) row M=L holds K{0..7,16..23}; lane L+16 holds K{8..15,24..31}
    int row = tm * 16 + (lane & 15);
    const unsigned short* ap = A + (size_t)row * 32;
    union { uint4 q[2]; v16bf v; } ua;
    ua.q[0] = *(const uint4*)(ap + half * 8);
    ua.q[1] = *(const uint4*)(ap + 16 + half * 8);

    int rbase = tm * 16 + (half << 3);
    #pragma unroll
    for (int j = 0; j < NTB; ++j) {
        int tn = tn0 + j;
        if (tn < NT) {                        // wave-uniform guard
            // B: pre-swizzled, 32 bytes per lane per tile
            const uint4* bp = (const uint4*)(Bp + ((size_t)(tn * 32 + lane)) * 16);
            union { uint4 q[2]; v16bf v; } ub;
            ub.q[0] = bp[0];
            ub.q[1] = bp[1];

            v8f c = {};
            c = __builtin_amdgcn_wmma_f32_16x16x32_bf16(false, ua.v, false, ub.v,
                                                        (short)0, c, false, false);

            int col = tn * 16 + (lane & 15);
            #pragma unroll
            for (int r = 0; r < 8; ++r)
                G[(size_t)(rbase + r) * NG + col] = c[r];
        }
    }
}

// ---------------- edge pass: msg + scatter-add ----------------
// one wave per edge; lane = output channel o; h_e broadcast via __shfl (ds).
__global__ void k_edge(const int* ei, const float* ea,
                       const float* w1, const float* b1,
                       const float* G, int NG, int dout,
                       float* agg, int E) {
    int lane = threadIdx.x & 31;
    int wpb  = blockDim.x >> 5;
    bool act = lane < dout;
    for (int e = blockIdx.x * wpb + (threadIdx.x >> 5); e < E;
         e += gridDim.x * wpb) {
        int src = ei[e], dst = ei[E + e];
        float eav = ea[e];
        float hk = eav * w1[lane] + b1[lane];
        hk = hk > 0.f ? hk : 0.f;
        const float* g = G + (size_t)src * NG;
        __builtin_prefetch(g, 0, 1);          // global_prefetch_b8 into cache
        float msg = 0.f;
        #pragma unroll
        for (int k = 0; k < 32; ++k) {
            float hb = __shfl(hk, k, 32);     // executed by all lanes (EXEC full)
            float gv = act ? g[k * dout + lane] : 0.f;
            msg += hb * gv;
        }
        if (act) {
            msg += g[32 * dout + lane];       // folded b2 term
            atomicAdd(&agg[(size_t)dst * dout + lane], msg);
        }
    }
}

// ---------------- node pass A: pre-BN value + BN stats ----------------
__global__ void k_nodeA(const float* G, int NG, int Cr,
                        const float* agg, const float* deg, const float* bias,
                        float* pre, float* bn, int N) {
    __shared__ float ssum[32], ssq[32];
    if (threadIdx.x < 32) { ssum[threadIdx.x] = 0.f; ssq[threadIdx.x] = 0.f; }
    __syncthreads();
    int idx = blockIdx.x * blockDim.x + threadIdx.x;
    if (idx < N * 32) {
        int n = idx >> 5, o = idx & 31;
        float p = G[(size_t)n * NG + Cr + o]
                + agg[idx] / fmaxf(deg[n], 1.0f) + bias[o];
        pre[idx] = p;
        atomicAdd(&ssum[o], p);               // ds_add_f32
        atomicAdd(&ssq[o], p * p);
    }
    __syncthreads();
    if (threadIdx.x < 32) {
        atomicAdd(&bn[threadIdx.x], ssum[threadIdx.x]);
        atomicAdd(&bn[32 + threadIdx.x], ssq[threadIdx.x]);
    }
}

// ---------------- node pass B: BN + relu (+residual), write f32 + bf16 ----------------
__global__ void k_nodeB(const float* pre, const float* bn,
                        const float* gamma, const float* beta,
                        const float* hin, int hasRes,
                        float* hout, unsigned short* hbf, int N) {
    int idx = blockIdx.x * blockDim.x + threadIdx.x;
    if (idx >= N * 32) return;
    int o = idx & 31;
    float inv  = 1.0f / (float)N;
    float mean = bn[o] * inv;
    float var  = bn[32 + o] * inv - mean * mean;
    float y = (pre[idx] - mean) * rsqrtf(var + 1e-5f) * gamma[o] + beta[o];
    float r = fmaxf(y, 0.f);
    if (hasRes) r += hin[idx];
    hout[idx] = r;
    hbf[idx]  = f2bf(r);
}

// ---------------- conv_out node pass (no BN) ----------------
__global__ void k_nodeOut(const float* G, int NG, int Cr,
                          const float* agg, const float* deg, const float* bias,
                          float* raw, int N) {
    int idx = blockIdx.x * blockDim.x + threadIdx.x;
    if (idx >= N * 3) return;
    int n = idx / 3, o = idx - n * 3;
    raw[idx] = G[(size_t)n * NG + Cr + o]
             + agg[idx] / fmaxf(deg[n], 1.0f) + bias[o];
}

// ---------------- scaler MLP + final output ----------------
__global__ void k_final(const float* h, const float* vemb, const float* x,
                        const float* raw,
                        const float* sw1, const float* sb1,
                        const float* sw2, const float* sb2,
                        float* out, int N) {
    int n = blockIdx.x * blockDim.x + threadIdx.x;
    if (n >= N) return;
    float ctx[40];
    #pragma unroll
    for (int i = 0; i < 32; ++i) ctx[i] = h[(size_t)n * 32 + i];
    #pragma unroll
    for (int i = 0; i < 8; ++i)  ctx[32 + i] = vemb[n * 8 + i];
    float z = sb2[0];
    for (int j = 0; j < 32; ++j) {
        float t = sb1[j];
        #pragma unroll
        for (int i = 0; i < 40; ++i) t += ctx[i] * sw1[i * 32 + j];
        t = t > 0.f ? t : 0.f;
        z += t * sw2[j];
    }
    float scale = 0.5f / (1.0f + expf(-z));
    #pragma unroll
    for (int o = 0; o < 3; ++o)
        out[n * 3 + o] = x[n * 10 + o] + raw[n * 3 + o] * scale;
}

// ---------------------------------------------------------------------------
extern "C" void kernel_launch(void* const* d_in, const int* in_sizes, int n_in,
                              void* d_out, int out_size, void* d_ws, size_t ws_size,
                              hipStream_t stream) {
    const int N = N_NODES, E = N_EDGES;

    // ---- inputs (setup_inputs() dict insertion order, recursive) ----
    const float* x    = (const float*)d_in[0];
    const int*   ei   = (const int*)  d_in[1];
    const float* ea   = (const float*)d_in[2];
    const float* vw1  = (const float*)d_in[3];
    const float* vb1  = (const float*)d_in[4];
    const float* vw2  = (const float*)d_in[5];
    const float* vb2  = (const float*)d_in[6];
    // conv layers: {w1,b1,w2,b2,root,bias[,gamma,beta]}
    const float* ci_w1 = (const float*)d_in[7];
    const float* ci_b1 = (const float*)d_in[8];
    const float* ci_w2 = (const float*)d_in[9];
    const float* ci_b2 = (const float*)d_in[10];
    const float* ci_rt = (const float*)d_in[11];
    const float* ci_bs = (const float*)d_in[12];
    const float* ci_g  = (const float*)d_in[13];
    const float* ci_be = (const float*)d_in[14];
    const float* h0_w1 = (const float*)d_in[15];
    const float* h0_b1 = (const float*)d_in[16];
    const float* h0_w2 = (const float*)d_in[17];
    const float* h0_b2 = (const float*)d_in[18];
    const float* h0_rt = (const float*)d_in[19];
    const float* h0_bs = (const float*)d_in[20];
    const float* h0_g  = (const float*)d_in[21];
    const float* h0_be = (const float*)d_in[22];
    const float* h1_w1 = (const float*)d_in[23];
    const float* h1_b1 = (const float*)d_in[24];
    const float* h1_w2 = (const float*)d_in[25];
    const float* h1_b2 = (const float*)d_in[26];
    const float* h1_rt = (const float*)d_in[27];
    const float* h1_bs = (const float*)d_in[28];
    const float* h1_g  = (const float*)d_in[29];
    const float* h1_be = (const float*)d_in[30];
    const float* co_w1 = (const float*)d_in[31];
    const float* co_b1 = (const float*)d_in[32];
    const float* co_w2 = (const float*)d_in[33];
    const float* co_b2 = (const float*)d_in[34];
    const float* co_rt = (const float*)d_in[35];
    const float* co_bs = (const float*)d_in[36];
    const float* sw1   = (const float*)d_in[37];
    const float* sb1   = (const float*)d_in[38];
    const float* sw2   = (const float*)d_in[39];
    const float* sb2   = (const float*)d_in[40];

    float* out = (float*)d_out;

    // ---- workspace carve-out (256B aligned) ----
    size_t off = 0;
    auto alloc = [&](size_t bytes) -> char* {
        off = (off + 255) & ~(size_t)255;
        char* p = (char*)d_ws + off;
        off += bytes;
        return p;
    };
    float*          vemb = (float*)alloc((size_t)N * 8 * 4);
    float*          hA   = (float*)alloc((size_t)N * 32 * 4);
    float*          hB   = (float*)alloc((size_t)N * 32 * 4);
    unsigned short* hbf  = (unsigned short*)alloc((size_t)N * 32 * 2);
    float*          G    = (float*)alloc((size_t)N * 1088 * 4);
    float*          agg  = (float*)alloc((size_t)N * 32 * 4);
    float*          deg  = (float*)alloc((size_t)N * 4);
    float*          pre  = (float*)alloc((size_t)N * 32 * 4);
    float*          bn   = (float*)alloc(64 * 4);
    float*          raw  = (float*)alloc((size_t)N * 3 * 4);
    unsigned short* BpIn = (unsigned short*)alloc((size_t)68 * 512 * 2);
    unsigned short* Bp0  = (unsigned short*)alloc((size_t)68 * 512 * 2);
    unsigned short* Bp1  = (unsigned short*)alloc((size_t)68 * 512 * 2);
    unsigned short* BpOut= (unsigned short*)alloc((size_t)7 * 512 * 2);
    (void)ws_size; (void)in_sizes; (void)n_in; (void)out_size;

    const int MT = N / 16;                 // 625 M-tiles
    const int NT32 = 68, NG32 = 1088;      // dout=32 layers: 34*32 cols
    const int NT3  = 7,  NG3  = 112;       // conv_out: 102 cols padded to 112

    // ---- degree (once) ----
    k_clear<<<64, 256, 0, stream>>>(deg, N);
    k_deg<<<(E + 255) / 256, 256, 0, stream>>>(ei, deg, E);

    // ---- voltage encoder + h0(bf16) ----
    k_venc<<<(N + 255) / 256, 256, 0, stream>>>(x, vw1, vb1, vw2, vb2, vemb, hbf, N);

    // ---- pack B matrices (bf16, WMMA lane layout) ----
    k_pack<<<(NT32 * 32 + 255) / 256, 256, 0, stream>>>(ci_w2, ci_b2, ci_rt, 18, 32, NT32, BpIn);
    k_pack<<<(NT32 * 32 + 255) / 256, 256, 0, stream>>>(h0_w2, h0_b2, h0_rt, 32, 32, NT32, Bp0);
    k_pack<<<(NT32 * 32 + 255) / 256, 256, 0, stream>>>(h1_w2, h1_b2, h1_rt, 32, 32, NT32, Bp1);
    k_pack<<<(NT3  * 32 + 255) / 256, 256, 0, stream>>>(co_w2, co_b2, co_rt, 32,  3, NT3,  BpOut);

    const int NGP32 = (NT32 + NTB - 1) / NTB;   // 17 strip-groups
    const int NGP3  = (NT3  + NTB - 1) / NTB;   // 2 strip-groups
    const int gemmBlocks32 = (MT * NGP32 + 7) / 8;
    const int gemmBlocks3  = (MT * NGP3  + 7) / 8;
    const int edgeBlocks   = (E + 7) / 8;       // 8 waves per 256-thread block
    const int nodeBlocks   = (N * 32 + 255) / 256;

    // ===== conv_in =====
    k_clear<<<512, 256, 0, stream>>>(agg, N * 32);
    k_clear<<<1, 64, 0, stream>>>(bn, 64);
    k_gemm<<<gemmBlocks32, 256, 0, stream>>>(hbf, BpIn, G, MT, NT32, NG32);
    k_edge<<<edgeBlocks, 256, 0, stream>>>(ei, ea, ci_w1, ci_b1, G, NG32, 32, agg, E);
    k_nodeA<<<nodeBlocks, 256, 0, stream>>>(G, NG32, 33 * 32, agg, deg, ci_bs, pre, bn, N);
    k_nodeB<<<nodeBlocks, 256, 0, stream>>>(pre, bn, ci_g, ci_be, hA, 0, hA, hbf, N);

    // ===== hidden[0] : in hA -> out hB (residual += hA) =====
    k_clear<<<512, 256, 0, stream>>>(agg, N * 32);
    k_clear<<<1, 64, 0, stream>>>(bn, 64);
    k_gemm<<<gemmBlocks32, 256, 0, stream>>>(hbf, Bp0, G, MT, NT32, NG32);
    k_edge<<<edgeBlocks, 256, 0, stream>>>(ei, ea, h0_w1, h0_b1, G, NG32, 32, agg, E);
    k_nodeA<<<nodeBlocks, 256, 0, stream>>>(G, NG32, 33 * 32, agg, deg, h0_bs, pre, bn, N);
    k_nodeB<<<nodeBlocks, 256, 0, stream>>>(pre, bn, h0_g, h0_be, hA, 1, hB, hbf, N);

    // ===== hidden[1] : in hB -> out hA (residual += hB) =====
    k_clear<<<512, 256, 0, stream>>>(agg, N * 32);
    k_clear<<<1, 64, 0, stream>>>(bn, 64);
    k_gemm<<<gemmBlocks32, 256, 0, stream>>>(hbf, Bp1, G, MT, NT32, NG32);
    k_edge<<<edgeBlocks, 256, 0, stream>>>(ei, ea, h1_w1, h1_b1, G, NG32, 32, agg, E);
    k_nodeA<<<nodeBlocks, 256, 0, stream>>>(G, NG32, 33 * 32, agg, deg, h1_bs, pre, bn, N);
    k_nodeB<<<nodeBlocks, 256, 0, stream>>>(pre, bn, h1_g, h1_be, hB, 1, hA, hbf, N);

    // ===== conv_out (no BN) : raw_disp = hA@root + agg/deg + bias =====
    k_clear<<<128, 256, 0, stream>>>(agg, N * 3);
    k_gemm<<<gemmBlocks3, 256, 0, stream>>>(hbf, BpOut, G, MT, NT3, NG3);
    k_edge<<<edgeBlocks, 256, 0, stream>>>(ei, ea, co_w1, co_b1, G, NG3, 3, agg, E);
    k_nodeOut<<<(N * 3 + 255) / 256, 256, 0, stream>>>(G, NG3, 33 * 3, agg, deg, co_bs, raw, N);

    // ===== scaler + output =====
    k_final<<<(N + 255) / 256, 256, 0, stream>>>(hA, vemb, x, raw, sw1, sb1, sw2, sb2, out, N);
}
